// LSTM_89945205113257
// MI455X (gfx1250) — matrix operand order repaired and checked by reading, not compiled, not run on hardware
//
#include <hip/hip_runtime.h>
#include <stdint.h>

#define B_   256
#define T_   512
#define H_   512
#define C_   10

typedef __bf16 v16bf __attribute__((ext_vector_type(16)));
typedef float  v8f   __attribute__((ext_vector_type(8)));

union FragU { v16bf v; int4 q[2]; };

// ---- CDNA5 async global->LDS copy (ASYNCcnt-tracked), per CDNA5 ISA
__device__ __forceinline__ void async_g2l_b128(uint32_t lds_off, uint64_t gaddr) {
  asm volatile("global_load_async_to_lds_b128 %0, %1, off"
               :: "v"(lds_off), "v"(gaddr) : "memory");
}
__device__ __forceinline__ void async_g2l_b128_dev(uint32_t lds_off, uint64_t gaddr) {
  asm volatile("global_load_async_to_lds_b128 %0, %1, off scope:SCOPE_DEV"
               :: "v"(lds_off), "v"(gaddr) : "memory");
}
__device__ __forceinline__ void wait_async0() {
  asm volatile("s_wait_asynccnt 0" ::: "memory");
}

// ---- one-time weight prep: Wt[gate][n][k] = (bf16) W_gate[k][n]  (B-operand, n-major)
__global__ void prep_weights_kernel(const float* __restrict__ wgh, const float* __restrict__ wih,
                                    const float* __restrict__ wfh, const float* __restrict__ woh,
                                    __bf16* __restrict__ Wt)
{
  int idx = blockIdx.x * blockDim.x + threadIdx.x;
  if (idx >= 4 * H_ * H_) return;
  int gate = idx >> 18;
  int rem  = idx & (H_ * H_ - 1);
  int n = rem >> 9, k = rem & (H_ - 1);
  const float* w = (gate == 0) ? wgh : (gate == 1) ? wih : (gate == 2) ? wfh : woh;
  Wt[idx] = (__bf16)w[k * H_ + n];
}

// ---- zero h ping-pong buffers + group barrier counters
__global__ void init_ws_kernel(uint32_t* __restrict__ hz, uint32_t* __restrict__ cnt)
{
  int idx = blockIdx.x * blockDim.x + threadIdx.x;
  if (idx < (2 * B_ * H_) / 2) hz[idx] = 0u;   // 2*256*512 bf16 as uints
  if (idx < 16) cnt[idx] = 0u;
}

// ---- main persistent recurrence kernel: 16 batch groups x 8 hidden-slice blocks
__global__ __launch_bounds__(128, 1)
void lstm_step_kernel(const float* __restrict__ x,
                      const __bf16* __restrict__ Wt,
                      const float* __restrict__ wgx, const float* __restrict__ wix,
                      const float* __restrict__ wfx, const float* __restrict__ wox,
                      const float* __restrict__ bgp, const float* __restrict__ bip,
                      const float* __restrict__ bfp, const float* __restrict__ bop,
                      __bf16* __restrict__ hglob, float* __restrict__ hfin,
                      unsigned* __restrict__ cnt)
{
  extern __shared__ char smem[];
  __bf16* wbuf = (__bf16*)smem;                                  // [256][520] bf16  (260 KB)
  __bf16* hbuf = (__bf16*)(smem + 256 * 520 * 2);                // [16][520]  bf16
  float*  cbuf = (float*)(smem + 256 * 520 * 2 + 16 * 520 * 2);  // [16][68]   f32
  float*  xbuf = cbuf + 16 * 68;                                 // [16]       f32

  const int grp  = blockIdx.x >> 3;       // batch tile (16 rows)
  const int wg   = blockIdx.x & 7;        // hidden slice (64 units)
  const int b0   = grp << 4;
  const int nh0  = wg << 6;
  const int tid  = threadIdx.x;
  const int lane = tid & 31;
  const int wave = tid >> 5;
  const int half = lane >> 4;
  const int l16  = lane & 15;
  const int nloc = (wave << 4) + l16;     // 0..63 hidden unit within slice
  const int ngl  = nh0 + nloc;            // global hidden index

  // ---- stage this block's weight slice into LDS once (reused for all 512 steps)
  {
    const uint32_t wb0 = (uint32_t)(uintptr_t)wbuf;
    for (int it = tid; it < 4 * 64 * 64; it += 128) {            // 16384 x b128
      int row  = it >> 6;                                         // gate*64 + r
      int c16  = it & 63;
      int gate = row >> 6, r = row & 63;
      uint64_t src = (uint64_t)(uintptr_t)(Wt + (((size_t)gate * H_ + nh0 + r) << 9))
                   + (uint32_t)(c16 << 4);
      async_g2l_b128(wb0 + (uint32_t)(row * 1040 + (c16 << 4)), src);
    }
  }
  for (int it = tid; it < 16 * 68; it += 128) cbuf[it] = 0.0f;    // c0 = 0
  wait_async0();
  __syncthreads();

  // loop-invariant rank-1 x-weights and biases for this hidden unit
  const float wxg = wgx[ngl], wxi = wix[ngl], wxf = wfx[ngl], wxo = wox[ngl];
  const float bgv = bgp[ngl], biv = bip[ngl], bfv = bfp[ngl], bov = bop[ngl];

  const uint32_t hb0 = (uint32_t)(uintptr_t)hbuf;
  unsigned* mycnt = cnt + grp;

  for (int t = 0; t < T_; ++t) {
    // ---- refill A-operand (h_t, 16 x 512 bf16) from ping-pong global buffer
    const __bf16* hsrc = hglob + (size_t)(t & 1) * (B_ * H_) + (size_t)b0 * H_;
    for (int it = tid; it < 16 * 64; it += 128) {
      int r = it >> 6, c16 = it & 63;
      uint64_t src = (uint64_t)(uintptr_t)(hsrc + ((size_t)r << 9)) + (uint32_t)(c16 << 4);
      async_g2l_b128_dev(hb0 + (uint32_t)(r * 1040 + (c16 << 4)), src);
    }
    if (tid < 16) xbuf[tid] = x[(size_t)(b0 + tid) * T_ + t];
    wait_async0();
    __syncthreads();

    // ---- 4 gate GEMM tiles: (16x512) @ (512x16) each, K-split into 16 WMMAs
    v8f acc[4] = {};
    const __bf16* arow = hbuf + l16 * 520 + half * 8;
    const __bf16* brow = wbuf + (size_t)nloc * 520 + half * 8;   // gate stride 64*520
    #pragma unroll
    for (int kt = 0; kt < 16; ++kt) {
      FragU a;
      a.q[0] = *(const int4*)(arow + kt * 32);
      a.q[1] = *(const int4*)(arow + kt * 32 + 16);
      #pragma unroll
      for (int gg = 0; gg < 4; ++gg) {
        FragU b;
        const __bf16* bptr = brow + gg * (64 * 520) + kt * 32;
        b.q[0] = *(const int4*)(bptr);
        b.q[1] = *(const int4*)(bptr + 16);
        acc[gg] = __builtin_amdgcn_wmma_f32_16x16x32_bf16(
            false, a.v, false, b.v, (short)0, acc[gg], false, false);
      }
    }

    // ---- elementwise cell update entirely in registers (reference: no tanh on g)
    __bf16* hdst = hglob + (size_t)((t + 1) & 1) * (B_ * H_);
    #pragma unroll
    for (int r = 0; r < 8; ++r) {
      int   m  = (half << 3) + r;               // C/D layout: M = r + 8*half
      float xv = xbuf[m];
      float pg = acc[0][r] + xv * wxg + bgv;
      float pi = acc[1][r] + xv * wxi + biv;
      float pf = acc[2][r] + xv * wxf + bfv;
      float po = acc[3][r] + xv * wxo + bov;
      float iv = 1.0f / (1.0f + __expf(-pi));
      float fv = 1.0f / (1.0f + __expf(-pf));
      float ov = 1.0f / (1.0f + __expf(-po));
      float cold = cbuf[m * 68 + nloc];
      float cn = pg * iv + cold * fv;
      float hv = cn * ov;
      cbuf[m * 68 + nloc] = cn;
      hdst[(size_t)(b0 + m) * H_ + ngl] = (__bf16)hv;
      if (t == T_ - 1) hfin[(size_t)(b0 + m) * H_ + ngl] = hv;
    }

    // ---- inter-block (per-group) step barrier: monotonic counter, agent scope
    __threadfence();
    __syncthreads();
    if (tid == 0) {
      __hip_atomic_fetch_add(mycnt, 1u, __ATOMIC_RELEASE, __HIP_MEMORY_SCOPE_AGENT);
      unsigned tgt = 8u * (unsigned)(t + 1);
      while (__hip_atomic_load(mycnt, __ATOMIC_ACQUIRE, __HIP_MEMORY_SCOPE_AGENT) < tgt)
        __builtin_amdgcn_s_sleep(2);
    }
    __syncthreads();
  }
}

// ---- tiny final projection: out[b,c] = h_T[b,:] . wph[c,:] + bp[c]
__global__ void proj_kernel(const float* __restrict__ hfin, const float* __restrict__ wph,
                            const float* __restrict__ bp, float* __restrict__ out)
{
  int i = blockIdx.x * blockDim.x + threadIdx.x;
  if (i >= B_ * C_) return;
  int b = i / C_, c = i % C_;
  float s = bp[c];
  const float* hp = hfin + (size_t)b * H_;
  const float* wp = wph + (size_t)c * H_;
  for (int k = 0; k < H_; ++k) s += hp[k] * wp[k];
  out[i] = s;
}

extern "C" void kernel_launch(void* const* d_in, const int* in_sizes, int n_in,
                              void* d_out, int out_size, void* d_ws, size_t ws_size,
                              hipStream_t stream)
{
  (void)in_sizes; (void)n_in; (void)out_size; (void)ws_size;
  const float* x   = (const float*)d_in[0];
  const float* wgx = (const float*)d_in[1];
  const float* wgh = (const float*)d_in[2];
  const float* bg  = (const float*)d_in[3];
  const float* wix = (const float*)d_in[4];
  const float* wih = (const float*)d_in[5];
  const float* bi  = (const float*)d_in[6];
  const float* wfx = (const float*)d_in[7];
  const float* wfh = (const float*)d_in[8];
  const float* bff = (const float*)d_in[9];
  const float* wox = (const float*)d_in[10];
  const float* woh = (const float*)d_in[11];
  const float* bo  = (const float*)d_in[12];
  const float* wph = (const float*)d_in[13];
  const float* bp  = (const float*)d_in[14];

  char* ws = (char*)d_ws;
  __bf16*   Wt    = (__bf16*)ws;                                        // 2 MB
  __bf16*   hglob = (__bf16*)(ws + (size_t)2 * 1024 * 1024);            // 512 KB (ping-pong)
  float*    hfin  = (float*)(ws + (size_t)2 * 1024 * 1024 + 512 * 1024);// 512 KB
  unsigned* cnt   = (unsigned*)(ws + (size_t)3 * 1024 * 1024);          // 64 B

  init_ws_kernel<<<512, 256, 0, stream>>>((uint32_t*)hglob, cnt);
  prep_weights_kernel<<<(4 * H_ * H_ + 255) / 256, 256, 0, stream>>>(wgh, wih, wfh, woh, Wt);

  size_t smem = (size_t)256 * 520 * 2 + 16 * 520 * 2 + 16 * 68 * 4 + 16 * 4; // ~287 KB
  lstm_step_kernel<<<128, 128, smem, stream>>>(x, Wt, wgx, wix, wfx, wox,
                                               bg, bi, bff, bo, hglob, hfin, cnt);

  proj_kernel<<<(B_ * C_ + 255) / 256, 256, 0, stream>>>(hfin, wph, bp, (float*)d_out);
}
